// DynamicRadiusChannelFusion_49770081026150
// MI455X (gfx1250) — matrix operand order repaired and checked by reading, not compiled, not run on hardware
//
#include <hip/hip_runtime.h>
#include <hip/hip_bf16.h>

typedef __attribute__((ext_vector_type(16))) _Float16 v16h;
typedef __attribute__((ext_vector_type(8)))  float    v8f;

#define B_   4
#define N_   16384
#define M_   2048
#define C_   128
#define OUT_ 256
#define K_   32
#define MAX_RADIUS 0.3f
#define BIG_ 1000000000.0f

// ---------------------------------------------------------------------------
// WMMA 16-bit A-fragment (16x32) lane/element mapping.
// Forward (lane,hi,e) -> ko:   j=e>>1; kb=(j<4)?2j:2j+8; ko=kb+8*hi+(e&1)
// Inverse ko -> (hi,e):        hi=(ko>>3)&1; base=ko-8*hi; e=base-((base&16)>>1)
// ---------------------------------------------------------------------------
__device__ __forceinline__ void frag_pos(int ko, int& hi, int& e) {
    hi = (ko >> 3) & 1;
    const int base = ko - (hi << 3);
    e = base - ((base & 16) >> 1);
}

// ---------------------------------------------------------------------------
// Kernel A: pack w1 (256x128) and w2 (128x128) into f16 WMMA B-fragment-major
// layout:  wf[((kc*8 + nt)*32 + lane)*16 + e] = w[(kc*32 + hi*8 + (e&7) + (e>>3)*16)*128 + nt*16 + n]
// (lane = hi*16 + n; per-lane fragment is 16 contiguous halves = 32 bytes)
// ---------------------------------------------------------------------------
__global__ void cvt_kernel(const float* __restrict__ w1,
                           const float* __restrict__ w2,
                           _Float16* __restrict__ w1f,
                           _Float16* __restrict__ w2f) {
    const int i = blockIdx.x * blockDim.x + threadIdx.x;
    const int e    = i & 15;
    const int lane = (i >> 4) & 31;
    const int nt   = (i >> 9) & 7;
    const int kc   = i >> 12;
    const int n  = lane & 15;
    const int hi = lane >> 4;
    const int kk = hi * 8 + (e & 7) + ((e >> 3) << 4);
    if (i < 2 * C_ * C_)
        w1f[i] = (_Float16)w1[(size_t)(kc * 32 + kk) * C_ + nt * 16 + n];
    if (i < C_ * C_)
        w2f[i] = (_Float16)w2[(size_t)(kc * 32 + kk) * C_ + nt * 16 + n];
}

// ---------------------------------------------------------------------------
// Kernel B: radius-masked KNN (one workgroup of 256 threads per query)
// ---------------------------------------------------------------------------
__global__ void knn_kernel(const float* __restrict__ points,
                           const int*   __restrict__ center_idx,
                           int*         __restrict__ idx_out) {
    __shared__ float sdist[N_];                 // 64 KB (of 320 KB/WGP)
    __shared__ unsigned long long wkey[8];

    const int q = blockIdx.x;                   // 0 .. B*M-1
    const int b = q / M_;
    const float* pts = points + (size_t)b * N_ * 3;

    const int ci = center_idx[q];
    const float cx = pts[ci * 3 + 0];
    const float cy = pts[ci * 3 + 1];
    const float cz = pts[ci * 3 + 2];
    const float csq = cx * cx + cy * cy + cz * cz;

    for (int i = threadIdx.x; i < N_; i += blockDim.x) {
        const float px = pts[i * 3 + 0];
        const float py = pts[i * 3 + 1];
        const float pz = pts[i * 3 + 2];
        const float psq   = px * px + py * py + pz * pz;
        const float inner = cx * px + cy * py + cz * pz;
        const float d2 = fmaxf(csq + psq - 2.0f * inner, 0.0f);
        const float d  = sqrtf(d2 + 1e-8f);
        sdist[i] = (d <= MAX_RADIUS) ? d : BIG_;   // masked >= 0
    }
    __syncthreads();

    const int lane = threadIdx.x & 31;
    const int wid  = threadIdx.x >> 5;          // wave32: 8 waves per block

    for (int j = 0; j < K_; ++j) {
        unsigned long long best = ~0ull;        // key = (fp32 bits << 32) | idx
        for (int i = threadIdx.x; i < N_; i += blockDim.x) {
            const unsigned int fb = __float_as_uint(sdist[i]);
            const unsigned long long key =
                ((unsigned long long)fb << 32) | (unsigned int)i;
            if (key < best) best = key;
        }
        #pragma unroll
        for (int off = 16; off > 0; off >>= 1) {
            unsigned long long o = __shfl_xor(best, off, 32);
            if (o < best) best = o;
        }
        if (lane == 0) wkey[wid] = best;
        __syncthreads();
        if (threadIdx.x == 0) {
            unsigned long long r = wkey[0];
            #pragma unroll
            for (int w = 1; w < 8; ++w)
                if (wkey[w] < r) r = wkey[w];
            const int win = (int)(r & 0xFFFFFFFFu);
            idx_out[q * K_ + j] = win;
            sdist[win] = __uint_as_float(0x7F7FFFFFu);  // FLT_MAX: exclude
        }
        __syncthreads();
    }
}

// ---------------------------------------------------------------------------
// Kernel C: edge gating MLP with WMMA (one workgroup of 128 threads per query)
// A-fragments live pre-swizzled in LDS (2x ds_load_b128 per fragment),
// B-fragments pre-packed in global ws (2x global_load_b128 per fragment).
// ---------------------------------------------------------------------------
__global__ void mlp_kernel(const float*    __restrict__ feats,
                           const int*      __restrict__ center_idx,
                           const _Float16* __restrict__ w1f,
                           const float*    __restrict__ b1,
                           const _Float16* __restrict__ w2f,
                           const float*    __restrict__ b2,
                           const float*    __restrict__ w3,
                           const float*    __restrict__ b3,
                           const int*      __restrict__ knn,
                           float*          __restrict__ out) {
    // comboSw: A-fragments of combo (32x256): [mt][kc=8][lane=32][e=16]
    __shared__ __align__(32) _Float16 comboSw[2 * 8 * 32 * 16];   // 16 KB
    // hSw: A-fragments of h (32x128): [mt][kc=4][lane=32][e=16]
    __shared__ __align__(32) _Float16 hSw[2 * 4 * 32 * 16];       //  8 KB
    __shared__ float nf32 [K_][C_];                               // 16 KB
    __shared__ float cw   [K_][C_];                               // 16 KB
    __shared__ float cf32 [C_];
    __shared__ float fused[C_];

    const int q   = blockIdx.x;
    const int b   = q / M_;
    const int tid = threadIdx.x;             // 128 threads = 4 waves
    const float* f = feats + (size_t)b * N_ * C_;

    // ---- gather center + neighbor features (store A-frag-swizzled) ----
    const int ci = center_idx[q];
    cf32[tid] = f[(size_t)ci * C_ + tid];
    for (int t = tid; t < K_ * C_; t += 128) {
        const int k = t >> 7, c = t & (C_ - 1);
        const float v = f[(size_t)knn[q * K_ + k] * C_ + c];
        nf32[k][c] = v;
        const int cc = C_ + c;                       // neighbor half of combo
        int hi, e; frag_pos(cc & 31, hi, e);
        const int lane = (hi << 4) | (k & 15);
        comboSw[((((k >> 4) << 3) + (cc >> 5)) * 32 + lane) * 16 + e] = (_Float16)v;
    }
    __syncthreads();
    for (int t = tid; t < K_ * C_; t += 128) {
        const int k = t >> 7, c = t & (C_ - 1);
        int hi, e; frag_pos(c & 31, hi, e);
        const int lane = (hi << 4) | (k & 15);
        comboSw[((((k >> 4) << 3) + (c >> 5)) * 32 + lane) * 16 + e] =
            (_Float16)cf32[c];
    }
    __syncthreads();

    const int lane = tid & 31;
    const int wid  = tid >> 5;
    const int rlo  = lane & 15;              // col (B / C-tile) within tile
    const int hi   = (lane >> 4) & 1;

    // ---- layer 1: (32 x 256) @ (256 x 128) -> relu -> hSw ----
    for (int t = wid; t < 16; t += 4) {      // uniform per wave: 4 tiles each
        const int mt = t & 1, nt = t >> 1;
        if (t + 4 < 16) {                    // prefetch next tile's B frags
            const int nnt = (t + 4) >> 1;
            __builtin_prefetch(&w1f[(size_t)(0 * 8 + nnt) * 512 + lane * 16], 0, 1);
        }
        v8f acc = {};
        #pragma unroll
        for (int kc = 0; kc < 8; ++kc) {
            const v16h a  = *(const v16h*)&comboSw[((mt * 8 + kc) * 32 + lane) * 16];
            const v16h bf = *(const v16h*)&w1f[(size_t)((kc * 8 + nt) * 32 + lane) * 16];
            acc = __builtin_amdgcn_wmma_f32_16x16x32_f16(
                false, a, false, bf, (short)0, acc, false, false);
        }
        const int   col = nt * 16 + rlo;     // N col of layer1 == K of layer2
        const float bb  = b1[col];
        int hi2, e2; frag_pos(col & 31, hi2, e2);
        #pragma unroll
        for (int r = 0; r < 8; ++r) {
            const int row = mt * 16 + hi * 8 + r;     // edge row
            float v = acc[r] + bb;
            v = v > 0.0f ? v : 0.0f;
            const int lane2 = (hi2 << 4) | (row & 15);
            hSw[((((row >> 4) << 2) + (col >> 5)) * 32 + lane2) * 16 + e2] =
                (_Float16)v;
        }
    }
    __syncthreads();

    // ---- layer 2: (32 x 128) @ (128 x 128) -> sigmoid -> cw ----
    for (int t = wid; t < 16; t += 4) {
        const int mt = t & 1, nt = t >> 1;
        v8f acc = {};
        #pragma unroll
        for (int kc = 0; kc < 4; ++kc) {
            const v16h a  = *(const v16h*)&hSw[((mt * 4 + kc) * 32 + lane) * 16];
            const v16h bf = *(const v16h*)&w2f[(size_t)((kc * 8 + nt) * 32 + lane) * 16];
            acc = __builtin_amdgcn_wmma_f32_16x16x32_f16(
                false, a, false, bf, (short)0, acc, false, false);
        }
        const int   col = nt * 16 + rlo;
        const float bb  = b2[col];
        #pragma unroll
        for (int r = 0; r < 8; ++r) {
            const float v = acc[r] + bb;
            cw[mt * 16 + hi * 8 + r][col] = 1.0f / (1.0f + expf(-v));
        }
    }
    __syncthreads();

    // ---- weighted mean over K + residual (deterministic order) ----
    {
        float s = 0.0f;
        #pragma unroll 4
        for (int k = 0; k < K_; ++k) s += nf32[k][tid] * cw[k][tid];
        fused[tid] = s * (1.0f / (float)K_) + cf32[tid];
    }
    __syncthreads();

    // ---- layer 3: (1 x 128) @ (128 x 256) -> relu ----
    for (int o = tid; o < OUT_; o += 128) {
        float s = b3[o];
        #pragma unroll 4
        for (int c = 0; c < C_; ++c) s += fused[c] * w3[(size_t)c * OUT_ + o];
        out[(size_t)q * OUT_ + o] = fmaxf(s, 0.0f);
    }
}

// ---------------------------------------------------------------------------
extern "C" void kernel_launch(void* const* d_in, const int* in_sizes, int n_in,
                              void* d_out, int out_size, void* d_ws, size_t ws_size,
                              hipStream_t stream) {
    (void)in_sizes; (void)n_in; (void)out_size; (void)ws_size;

    const float* points     = (const float*)d_in[0];
    const float* feats      = (const float*)d_in[1];
    const int*   center_idx = (const int*)  d_in[2];
    const float* w1         = (const float*)d_in[3];
    const float* b1         = (const float*)d_in[4];
    const float* w2         = (const float*)d_in[5];
    const float* b2         = (const float*)d_in[6];
    const float* w3         = (const float*)d_in[7];
    const float* b3         = (const float*)d_in[8];

    float* out     = (float*)d_out;                       // (B,M,OUT) f32
    int*   idx_out = (int*)((float*)d_out + (size_t)B_ * M_ * OUT_); // (B,M,K) i32

    _Float16* w1f = (_Float16*)d_ws;                      // 64 KB, frag-major
    _Float16* w2f = (_Float16*)((char*)d_ws + 2 * C_ * C_ * sizeof(_Float16));

    // 1) weight conversion + fragment packing
    cvt_kernel<<<(2 * C_ * C_ + 255) / 256, 256, 0, stream>>>(w1, w2, w1f, w2f);

    // 2) radius-masked KNN
    knn_kernel<<<B_ * M_, 256, 0, stream>>>(points, center_idx, idx_out);

    // 3) edge gating MLP (WMMA)
    mlp_kernel<<<B_ * M_, 128, 0, stream>>>(feats, center_idx, w1f, b1, w2f, b2,
                                            w3, b3, idx_out, out);
}
